// BidirLayer_14568529068707
// MI455X (gfx1250) — compile-verified, matrix operand order checked by reading, not run on hardware
//
#include <hip/hip_runtime.h>
#include <cmath>

// ---------------------------------------------------------------------------
// Bidirectional GRU layer for MI455X (gfx1250, wave32, WMMA).
//   T=512, B=64, I=H=512.  gate order: r, z, n  (PyTorch GRUCell math)
//
// Plan:
//   1) convert inp / Wih / Whh to bf16 (WMMA operand type, f32 accumulate)
//   2) gi[dir] = X @ Wih^T + bih  as one large WMMA GEMM (time-batched)
//   3) persistent recurrent kernel: grid=2 (one WG/direction, 32 waves),
//      h in LDS (bf16 for WMMA-A via ds_load_b128, f32 for z*h),
//      Whh streamed from L2, gh scratch bounced through L2.
// ---------------------------------------------------------------------------

typedef __bf16 bf16;
typedef bf16  bf16x8  __attribute__((ext_vector_type(8)));
typedef bf16  bf16x16 __attribute__((ext_vector_type(16)));
typedef float floatx8 __attribute__((ext_vector_type(8)));

union BF16x16 { bf16x16 v; bf16x8 h[2]; };

static constexpr int T  = 512;
static constexpr int Bb = 64;
static constexpr int I  = 512;
static constexpr int H  = 512;
static constexpr int G  = 3 * H;      // 1536
static constexpr int TB = T * Bb;     // 32768

// --------------------------- fragment loader -------------------------------
// 16-bit A/B fragment for v_wmma_f32_16x16x32_bf16. "row" is the M row (A)
// or N column (B); K is contiguous in memory (row-major X, row-major W which
// is column-major W^T). Per ISA 7.12.2: lanes 0-15 hold K = kb+[0..7] and
// kb+16..23; lanes 16-31 hold K = kb+8..15 and kb+24..31, packed 2/VGPR.
__device__ inline bf16x16 load_frag(const bf16* __restrict__ base, int stride,
                                    int row, int kb, int lane) {
  const bf16* p = base + (size_t)row * stride + kb + ((lane >> 4) << 3);
  BF16x16 f;
  f.h[0] = *(const bf16x8*)(p);
  f.h[1] = *(const bf16x8*)(p + 16);
  return f.v;
}

// --------------------------- f32 -> bf16 cast ------------------------------
__global__ void cast_f32_bf16(const float* __restrict__ in,
                              bf16* __restrict__ out, int n) {
  int i = blockIdx.x * blockDim.x + threadIdx.x;
  if (i < n) out[i] = (bf16)in[i];
}

// --------------------- time-batched input projection -----------------------
// gi[row][col] = X[row][:] . W[col][:] + bih[col]
// Workgroup: 8 waves -> 64(M) x 256(N) tile. Wave: 32(M) x 64(N) = 2x4 WMMA.
__global__ __launch_bounds__(256)
void gi_gemm(const bf16* __restrict__ X,      // [TB][I] bf16
             const bf16* __restrict__ W,      // [G][I]  bf16 (Wih row-major)
             const float* __restrict__ bih,   // [G]
             float* __restrict__ gi) {        // [TB][G] f32
  const int lane = threadIdx.x & 31;
  const int wave = threadIdx.x >> 5;
  const int Mbase = blockIdx.x * 64 + (wave >> 2) * 32;
  const int Nbase = blockIdx.y * 256 + (wave & 3) * 64;

  floatx8 c[2][4] = {};
  for (int kk = 0; kk < I / 32; ++kk) {
    const int kb = kk * 32;
    if (kk + 1 < I / 32) {  // stream next weight block (global_prefetch_b8)
      __builtin_prefetch(W + (size_t)(Nbase + (lane & 15)) * I + kb + 32, 0, 0);
    }
    bf16x16 a[2], b[4];
#pragma unroll
    for (int mi = 0; mi < 2; ++mi)
      a[mi] = load_frag(X, I, Mbase + mi * 16 + (lane & 15), kb, lane);
#pragma unroll
    for (int nj = 0; nj < 4; ++nj)
      b[nj] = load_frag(W, I, Nbase + nj * 16 + (lane & 15), kb, lane);
#pragma unroll
    for (int mi = 0; mi < 2; ++mi)
#pragma unroll
      for (int nj = 0; nj < 4; ++nj)
        c[mi][nj] = __builtin_amdgcn_wmma_f32_16x16x32_bf16(
            false, a[mi], false, b[nj], (short)0, c[mi][nj], false, false);
  }

  // C layout (7.12.2): VGPR r -> M = r + (lane<16 ? 0 : 8), N = lane&15
#pragma unroll
  for (int mi = 0; mi < 2; ++mi)
#pragma unroll
    for (int nj = 0; nj < 4; ++nj) {
      const int col = Nbase + nj * 16 + (lane & 15);
      const float bb = bih[col];
      float* o = gi + (size_t)(Mbase + mi * 16 + ((lane >> 4) << 3)) * G + col;
#pragma unroll
      for (int r = 0; r < 8; ++r)
        o[(size_t)r * G] = c[mi][nj][r] + bb;
    }
}

// ------------------------------ recurrence ---------------------------------
// grid = {2} (dir = blockIdx.x), 1024 threads = 32 wave32 on one WGP.
// LDS: h as f32 (128 KB) + bf16 (64 KB) = 192 KB of the WGP's 320 KB.
__global__ __launch_bounds__(1024)
void gru_recurrent(const float* __restrict__ gi_f, const float* __restrict__ gi_b,
                   const bf16*  __restrict__ Whh_f, const bf16* __restrict__ Whh_b,
                   const float* __restrict__ bhh_f, const float* __restrict__ bhh_b,
                   const float* __restrict__ h0_f,  const float* __restrict__ h0_b,
                   float* __restrict__ scr_f, float* __restrict__ scr_b,
                   float* __restrict__ out,   // [T][B][2H]
                   float* __restrict__ hT) {  // [2][B][H]
  const int dir = blockIdx.x;
  const float* gi  = dir ? gi_b  : gi_f;
  const bf16*  Whh = dir ? Whh_b : Whh_f;
  const float* bhh = dir ? bhh_b : bhh_f;
  const float* h0  = dir ? h0_b  : h0_f;
  float* scr = dir ? scr_b : scr_f;      // [B][G] gh scratch (L2-resident)
  float* outc = out + (size_t)dir * H;   // column offset for this direction

  extern __shared__ char smem[];
  float* hf = (float*)smem;                              // [B][H] f32
  bf16*  hb = (bf16*)(smem + (size_t)Bb * H * sizeof(float));  // [B][H] bf16

  const int tid  = threadIdx.x;
  const int lane = tid & 31;
  const int wave = tid >> 5;

  for (int i = tid; i < Bb * H; i += 1024) {
    float v = h0[i];
    hf[i] = v;
    hb[i] = (bf16)v;
  }
  __syncthreads();

  for (int step = 0; step < T; ++step) {
    const int t = dir ? (T - 1 - step) : step;
    const float* git = gi + (size_t)t * Bb * G;

    // ---- phase 1: gh = h @ Whh^T + bhh  -> scr ----
    // wave owns 3 N-strips of 16 cols; each strip: 4 M-tiles, K-loop of 16.
    for (int j = 0; j < 3; ++j) {
      const int Ncol = wave * 48 + j * 16;
      floatx8 c[4] = {};
      for (int kk = 0; kk < H / 32; ++kk) {
        const int kb = kk * 32;
        if (kk + 1 < H / 32) {
          __builtin_prefetch(Whh + (size_t)(Ncol + (lane & 15)) * H + kb + 32, 0, 3);
        }
        bf16x16 bfrg = load_frag(Whh, H, Ncol + (lane & 15), kb, lane);
#pragma unroll
        for (int mi = 0; mi < 4; ++mi) {
          bf16x16 afrg = load_frag(hb, H, mi * 16 + (lane & 15), kb, lane);
          c[mi] = __builtin_amdgcn_wmma_f32_16x16x32_bf16(
              false, afrg, false, bfrg, (short)0, c[mi], false, false);
        }
      }
      const int col = Ncol + (lane & 15);
      const float bb = bhh[col];
#pragma unroll
      for (int mi = 0; mi < 4; ++mi) {
        float* o = scr + (size_t)(mi * 16 + ((lane >> 4) << 3)) * G + col;
#pragma unroll
        for (int r = 0; r < 8; ++r)
          o[(size_t)r * G] = c[mi][r] + bb;
      }
    }
    __threadfence();   // make scr stores visible before phase 2 reads
    __syncthreads();

    // ---- phase 2: gates + state update + output row ----
    for (int i = tid; i < Bb * H; i += 1024) {
      const int b = i >> 9;          // H == 512
      const int n = i & (H - 1);
      const float* gib = git + (size_t)b * G;
      const float* sb  = scr + (size_t)b * G;
      const float sr = sb[n], sz = sb[H + n], sn = sb[2 * H + n];
      const float r  = 1.0f / (1.0f + __expf(-(gib[n] + sr)));
      const float z  = 1.0f / (1.0f + __expf(-(gib[H + n] + sz)));
      const float nn = tanhf(gib[2 * H + n] + r * sn);
      const float hv = (1.0f - z) * nn + z * hf[i];
      hf[i] = hv;
      hb[i] = (bf16)hv;
      outc[((size_t)t * Bb + b) * (2 * H) + n] = hv;
    }
    __syncthreads();   // h updated before next step's WMMA reads
  }

  for (int i = tid; i < Bb * H; i += 1024)
    hT[(size_t)dir * Bb * H + i] = hf[i];
}

// ------------------------------- launcher ----------------------------------
extern "C" void kernel_launch(void* const* d_in, const int* in_sizes, int n_in,
                              void* d_out, int out_size, void* d_ws, size_t ws_size,
                              hipStream_t stream) {
  (void)in_sizes; (void)n_in; (void)out_size; (void)ws_size;
  const float* inp  = (const float*)d_in[0];
  const float* h0f  = (const float*)d_in[1];
  const float* h0b  = (const float*)d_in[2];
  const float* Wihf = (const float*)d_in[3];
  const float* Whhf = (const float*)d_in[4];
  const float* bihf = (const float*)d_in[5];
  const float* bhhf = (const float*)d_in[6];
  const float* Wihb = (const float*)d_in[7];
  const float* Whhb = (const float*)d_in[8];
  const float* bihb = (const float*)d_in[9];
  const float* bhhb = (const float*)d_in[10];
  float* out = (float*)d_out;

  char* ws = (char*)d_ws;
  auto carve = [&](size_t bytes) {
    char* p = ws;
    ws += (bytes + 255) & ~(size_t)255;
    return p;
  };
  bf16*  Xb    = (bf16*)carve((size_t)TB * I * sizeof(bf16));   //  32 MB
  bf16*  WihfB = (bf16*)carve((size_t)G * I * sizeof(bf16));
  bf16*  WihbB = (bf16*)carve((size_t)G * I * sizeof(bf16));
  bf16*  WhhfB = (bf16*)carve((size_t)G * H * sizeof(bf16));
  bf16*  WhhbB = (bf16*)carve((size_t)G * H * sizeof(bf16));
  float* giF   = (float*)carve((size_t)TB * G * sizeof(float)); // 192 MB
  float* giB   = (float*)carve((size_t)TB * G * sizeof(float)); // 192 MB
  float* scrF  = (float*)carve((size_t)Bb * G * sizeof(float));
  float* scrB  = (float*)carve((size_t)Bb * G * sizeof(float));

  // 1) bf16 conversions
  {
    int n = TB * I;
    cast_f32_bf16<<<(n + 255) / 256, 256, 0, stream>>>(inp, Xb, n);
    n = G * I;
    cast_f32_bf16<<<(n + 255) / 256, 256, 0, stream>>>(Wihf, WihfB, n);
    cast_f32_bf16<<<(n + 255) / 256, 256, 0, stream>>>(Wihb, WihbB, n);
    n = G * H;
    cast_f32_bf16<<<(n + 255) / 256, 256, 0, stream>>>(Whhf, WhhfB, n);
    cast_f32_bf16<<<(n + 255) / 256, 256, 0, stream>>>(Whhb, WhhbB, n);
  }

  // 2) time-batched input projections (both directions)
  {
    dim3 grid(TB / 64, G / 256);
    gi_gemm<<<grid, 256, 0, stream>>>(Xb, WihfB, bihf, giF);
    gi_gemm<<<grid, 256, 0, stream>>>(Xb, WihbB, bihb, giB);
  }

  // 3) persistent recurrence, both directions concurrently (grid = 2)
  {
    const size_t lds = (size_t)Bb * H * (sizeof(float) + sizeof(bf16)); // 192 KB
    hipFuncSetAttribute((const void*)gru_recurrent,
                        hipFuncAttributeMaxDynamicSharedMemorySize, (int)lds);
    float* hT = out + (size_t)T * Bb * 2 * H;
    gru_recurrent<<<2, 1024, lds, stream>>>(giF, giB, WhhfB, WhhbB,
                                            bhhf, bhhb, h0f, h0b,
                                            scrF, scrB, out, hT);
  }
}
